// BayesLSTM_21921513079501
// MI455X (gfx1250) — compile-verified, matrix operand order checked by reading, not run on hardware
//
#include <hip/hip_runtime.h>
#include <hip/hip_bf16.h>

// BayesLSTM for MI455X (gfx1250, wave32, WMMA).
// Phase 1: gates_x = (seq*mask_x) @ W_ih^T + b_ih + b_hh
//          LDS-staged bf16 WMMA GEMM, block tile 64(M) x 128(N), K-chunk 64.
// Phase 2: persistent 64-WG kernel; W_hh resident in LDS (bf16), c in registers,
//          wmma recurrent GEMM + fused LSTM cell + L2-atomic grid barrier per step.

typedef __attribute__((ext_vector_type(16))) __bf16 v16bf;
typedef __attribute__((ext_vector_type(8)))  __bf16 v8bf;
typedef __attribute__((ext_vector_type(8)))  float  v8f;

#define T_STEPS 512
#define B_SZ    32
#define I_SZ    1024
#define H_SZ    1024
#define G4H     4096
#define NWG     64          // phase-2 workgroups (persistent)

__device__ __forceinline__ __bf16 f2bf(float x) { return (__bf16)x; }

__device__ __forceinline__ v16bf make_frag(v8bf lo, v8bf hi) {
    return __builtin_shufflevector(lo, hi, 0,1,2,3,4,5,6,7,8,9,10,11,12,13,14,15);
}

// Fragment from a row-major bf16 LDS panel (ISA 16-bit A/B layout:
// lane -> row = lane&15, K = {khalf*8..+7} U {16+khalf*8..+7}).
__device__ __forceinline__ v16bf lds_frag(const __bf16* base, int row, int stride,
                                          int kofs, int khalf) {
    const __bf16* p = base + (size_t)row * stride + kofs + khalf * 8;
    return make_frag(*(const v8bf*)p, *(const v8bf*)(p + 16));
}

// ---------------------------------------------------------------------------
// Phase 1: gates_x[T*B, 4H] = (seq * mask_x) @ W_ih^T + b_ih + b_hh
// 256 threads / 8 waves. Block tile M=64, N=128. A and B staged in LDS as bf16
// per 64-wide K chunk; each wave owns one 16-wide N column and 4 M tiles.
// ---------------------------------------------------------------------------
__global__ __launch_bounds__(256) void bayeslstm_gatesx_gemm(
    const float* __restrict__ seq, const float* __restrict__ mask_x,
    const float* __restrict__ W_ih, const float* __restrict__ b_ih,
    const float* __restrict__ b_hh, float* __restrict__ gates_x)
{
    __shared__ __bf16 aA[64 * 64];    //  8 KB: masked seq tile (bf16)
    __shared__ __bf16 aB[128 * 64];   // 16 KB: W_ih tile (bf16)

    const int tid   = threadIdx.x;
    const int lane  = tid & 31;
    const int wave  = tid >> 5;
    const int n15   = lane & 15;
    const int khalf = lane >> 4;
    const int row_base  = blockIdx.y * 64;    // rows of [T*B]
    const int col_block = blockIdx.x * 128;   // cols of [4H]

    // Staging coordinates (fixed per thread)
    const int arow = tid >> 2, ak0 = (tid & 3) * 16;   // A: 16 elems/thread
    const int brow = tid >> 1, bk0 = (tid & 1) * 32;   // B: 32 elems/thread
    const float* a_seq = seq    + (size_t)(row_base + arow) * I_SZ + ak0;
    const float* a_msk = mask_x + (size_t)((row_base + arow) & (B_SZ - 1)) * I_SZ + ak0;
    const float* b_src = W_ih   + (size_t)(col_block + brow) * I_SZ + bk0;

    v8f acc[4] = {};
    for (int kb = 0; kb < I_SZ; kb += 64) {
        __syncthreads();   // previous chunk's fragment reads done
        // Stage A = seq * mask_x (fp32 -> bf16)
        #pragma unroll
        for (int i = 0; i < 16; ++i)
            aA[arow * 64 + ak0 + i] = f2bf(a_seq[kb + i] * a_msk[kb + i]);
        // Stage B = W_ih
        #pragma unroll
        for (int i = 0; i < 32; ++i)
            aB[brow * 64 + bk0 + i] = f2bf(b_src[kb + i]);
        // Prefetch next K chunk (global_prefetch_b8)
        if (kb + 64 < I_SZ) {
            __builtin_prefetch(a_seq + kb + 64, 0, 1);
            __builtin_prefetch(b_src + kb + 64, 0, 1);
        }
        __syncthreads();

        #pragma unroll
        for (int kk = 0; kk < 64; kk += 32) {
            v16bf b = lds_frag(aB, wave * 16 + n15, 64, kk, khalf);
            #pragma unroll
            for (int mt = 0; mt < 4; ++mt) {
                v16bf a = lds_frag(aA, mt * 16 + n15, 64, kk, khalf);
                acc[mt] = __builtin_amdgcn_wmma_f32_16x16x32_bf16(
                    false, a, false, b, (short)0, acc[mt], false, false);
            }
        }
    }

    const int   n    = col_block + wave * 16 + n15;   // C layout: n = lane&15
    const float bias = b_ih[n] + b_hh[n];
    const int   moff = 8 * khalf;                     // C layout: m = r + 8*(lane>>4)
    #pragma unroll
    for (int mt = 0; mt < 4; ++mt) {
        #pragma unroll
        for (int r = 0; r < 8; ++r)
            gates_x[(size_t)(row_base + mt * 16 + moff + r) * G4H + n] = acc[mt][r] + bias;
    }
}

// ---------------------------------------------------------------------------
// Phase 2: persistent recurrent scan. 64 WGs x 256 threads (8 waves).
// WG owns hidden units [hbase, hbase+16); its 64 W_hh rows (4 gates x 16)
// live in LDS as bf16 for all 512 steps. h double-buffered in global ws.
// Grid barrier: L2 atomic counter, release (threadfence) / acquire pattern.
// ---------------------------------------------------------------------------
__global__ __launch_bounds__(256) void bayeslstm_scan(
    const float* __restrict__ gates_x, const float* __restrict__ c0_in,
    const float* __restrict__ mask_h,  const float* __restrict__ W_hh,
    float* h_buf /* [2][B*H] */, unsigned* counter, float* __restrict__ out)
{
    extern __shared__ char smem[];
    __bf16* w_lds = (__bf16*)smem;                                   // [64][1024] bf16, 128 KB
    __bf16* h_lds = (__bf16*)(smem + 64 * 1024 * 2);                 // [32][1024] bf16,  64 KB
    float*  g_lds = (float*)(smem + 64 * 1024 * 2 + 32 * 1024 * 2);  // [32][64]   f32,    8 KB

    const int tid   = threadIdx.x;
    const int lane  = tid & 31;
    const int wave  = tid >> 5;
    const int n15   = lane & 15;
    const int khalf = lane >> 4;
    const int hbase = blockIdx.x * 16;
    const int mt    = wave & 1;      // M tile (batch 0..15 / 16..31)
    const int nt    = wave >> 1;     // N tile == gate index (i,f,g,o)

    // Stage W_hh slice -> LDS bf16. LDS row rr = gate*16 + r  <->  W_hh[gate*H + hbase + r]
    for (int idx = tid; idx < 64 * 1024; idx += 256) {
        const int rr = idx >> 10, k = idx & 1023;
        const int grow = (rr >> 4) * H_SZ + hbase + (rr & 15);
        w_lds[idx] = f2bf(W_hh[(size_t)grow * H_SZ + k]);
    }

    // Register-resident cell state: pairs p = 2*tid+e -> (m = p>>4, j = p&15)
    float cst[2];
    #pragma unroll
    for (int e = 0; e < 2; ++e) {
        const int p = 2 * tid + e;
        cst[e] = c0_in[(size_t)(p >> 4) * H_SZ + hbase + (p & 15)];
    }

    for (int t = 0; t < T_STEPS; ++t) {
        const volatile float* hread = h_buf + (size_t)(t & 1) * (B_SZ * H_SZ);
        float* hwrite = h_buf + (size_t)((t + 1) & 1) * (B_SZ * H_SZ);

        // (h * mask_h) -> bf16 LDS  (volatile: h written by other WGPs via L2)
        for (int idx = tid; idx < B_SZ * H_SZ; idx += 256)
            h_lds[idx] = f2bf(hread[idx] * mask_h[idx]);
        __syncthreads();

        // Recurrent GEMM tile: C[16,16] over K=1024 with bf16 WMMA
        const __bf16* arow = h_lds + (size_t)(mt * 16 + n15) * 1024;
        const __bf16* brow = w_lds + (size_t)(nt * 16 + n15) * 1024;
        v8f c = {};
        for (int kb = 0; kb < H_SZ; kb += 32) {
            v16bf a = make_frag(*(const v8bf*)(arow + kb + khalf * 8),
                                *(const v8bf*)(arow + kb + 16 + khalf * 8));
            v16bf b = make_frag(*(const v8bf*)(brow + kb + khalf * 8),
                                *(const v8bf*)(brow + kb + 16 + khalf * 8));
            c = __builtin_amdgcn_wmma_f32_16x16x32_bf16(false, a, false, b, (short)0, c, false, false);
        }
        #pragma unroll
        for (int r = 0; r < 8; ++r)
            g_lds[(mt * 16 + 8 * khalf + r) * 64 + nt * 16 + n15] = c[r];
        __syncthreads();

        // Fused LSTM cell (PyTorch gate order i,f,g,o)
        #pragma unroll
        for (int e = 0; e < 2; ++e) {
            const int p = 2 * tid + e;
            const int m = p >> 4, j = p & 15;
            const float* gx = gates_x + ((size_t)t * B_SZ + m) * G4H + hbase + j;
            const float gi = gx[0]          + g_lds[m * 64 + 0 * 16 + j];
            const float gf = gx[1 * H_SZ]   + g_lds[m * 64 + 1 * 16 + j];
            const float gg = gx[2 * H_SZ]   + g_lds[m * 64 + 2 * 16 + j];
            const float go = gx[3 * H_SZ]   + g_lds[m * 64 + 3 * 16 + j];
            const float si = 1.f / (1.f + __expf(-gi));
            const float sf = 1.f / (1.f + __expf(-gf));
            const float so = 1.f / (1.f + __expf(-go));
            const float cn = sf * cst[e] + si * tanhf(gg);
            const float hn = so * tanhf(cn);
            cst[e] = cn;
            const size_t hidx = (size_t)m * H_SZ + hbase + j;
            hwrite[hidx] = hn;
            out[(size_t)t * (B_SZ * H_SZ) + hidx] = hn;
            if (t == T_STEPS - 1) {
                out[(size_t)T_STEPS * B_SZ * H_SZ + hidx] = hn;                  // final h
                out[(size_t)T_STEPS * B_SZ * H_SZ + B_SZ * H_SZ + hidx] = cn;    // final c
            }
        }

        // Grid-wide barrier: release stores, bump counter, spin (s_sleep) until all WGs arrive.
        __threadfence();
        __syncthreads();
        if (tid == 0) {
            atomicAdd(counter, 1u);
            const unsigned target = (unsigned)NWG * (unsigned)(t + 1);
            while (atomicAdd(counter, 0u) < target)
                __builtin_amdgcn_s_sleep(2);
        }
        __syncthreads();
        __threadfence();
    }
}

// ---------------------------------------------------------------------------
extern "C" void kernel_launch(void* const* d_in, const int* in_sizes, int n_in,
                              void* d_out, int out_size, void* d_ws, size_t ws_size,
                              hipStream_t stream)
{
    const float* seq    = (const float*)d_in[0];
    const float* h0     = (const float*)d_in[1];
    const float* c0     = (const float*)d_in[2];
    const float* mask_x = (const float*)d_in[3];
    const float* mask_h = (const float*)d_in[4];
    const float* W_ih   = (const float*)d_in[5];
    const float* W_hh   = (const float*)d_in[6];
    const float* b_ih   = (const float*)d_in[7];
    const float* b_hh   = (const float*)d_in[8];
    float* out = (float*)d_out;

    // Workspace layout: [h double buffer][counter pad][gates_x]
    char* ws = (char*)d_ws;
    float*    h_buf   = (float*)ws;                                             // 2*32*1024 f32
    unsigned* counter = (unsigned*)(ws + 2 * B_SZ * H_SZ * sizeof(float));      // 512 B pad
    float*    gates_x = (float*)(ws + 2 * B_SZ * H_SZ * sizeof(float) + 512);   // 256 MB

    hipMemsetAsync(counter, 0, 512, stream);
    hipMemcpyAsync(h_buf, h0, B_SZ * H_SZ * sizeof(float), hipMemcpyDeviceToDevice, stream);

    dim3 g1(G4H / 128, (T_STEPS * B_SZ) / 64);
    bayeslstm_gatesx_gemm<<<g1, 256, 0, stream>>>(seq, mask_x, W_ih, b_ih, b_hh, gates_x);

    const size_t lds_bytes = 64 * 1024 * 2 + 32 * 1024 * 2 + 32 * 64 * 4; // 204800 B < 320 KB WGP LDS
    bayeslstm_scan<<<NWG, 256, lds_bytes, stream>>>(gates_x, c0, mask_h, W_hh, h_buf, counter, out);
}
